// ObservableEqProp_46084999086196
// MI455X (gfx1250) — matrix operand order repaired and checked by reading, not compiled, not run on hardware
//
#include <hip/hip_runtime.h>
#include <hip/hip_bf16.h>
#include <math.h>

// ---------------------------------------------------------------------------
// Problem constants (from reference): B=4096, IN=1024, H=2048, OUT=1024, L=3
// ---------------------------------------------------------------------------
constexpr int BATCH = 4096;
constexpr int INF   = 1024;
constexpr int HID   = 2048;
constexpr int OUTF  = 1024;
constexpr int NLAY  = 3;
constexpr int STEPS = 30;          // fixed python scalar in setup_inputs()
constexpr float ALPHA = 0.5f;
constexpr float EPSV  = 1e-12f;

typedef __bf16 v16bf __attribute__((ext_vector_type(16)));
typedef __bf16 v8bf  __attribute__((ext_vector_type(8)));
typedef float  v8f   __attribute__((ext_vector_type(8)));

union FragAB { v16bf v; v8bf h[2]; };

// ---------------------------------------------------------------------------
// CDNA5 async tile staging: global -> LDS DMA, tracked by ASYNCcnt.
// Each call moves 2 x (32 lanes x 16B) with per-lane LDS scatter addresses.
// ---------------------------------------------------------------------------
__device__ __forceinline__ void async_copy_32B(unsigned lds_addr, const __bf16* gaddr) {
  asm volatile(
      "global_load_async_to_lds_b128 %0, %1, off\n\t"
      "global_load_async_to_lds_b128 %0, %1, off offset:16"
      :: "v"(lds_addr), "v"(gaddr)
      : "memory");
}

__device__ __forceinline__ void wait_async0() {
  asm volatile("s_wait_asynccnt 0" ::: "memory");
}

// Low 32 bits of a flat pointer to a __shared__ object == LDS byte offset
// (flat->LDS aperture mapping uses addr[31:0]).
__device__ __forceinline__ unsigned lds_off(const void* p) {
  return (unsigned)(unsigned long long)p;
}

// ---------------------------------------------------------------------------
// Spectral norm: one power-iteration step per layer.
//   v = normalize(W^T u); sigma = ||W v||^2 / (||W v|| + eps)
// One block (256 threads) per layer. Cost: 2 * 4M MACs per layer -> negligible.
// ---------------------------------------------------------------------------
__global__ __launch_bounds__(256)
void sigma_kernel(const float* __restrict__ Ws, const float* __restrict__ us,
                  float* __restrict__ inv_sigma) {
  const int l = blockIdx.x;
  const float* W = Ws + (size_t)l * HID * HID;
  const float* u = us + (size_t)l * HID;
  __shared__ float v[HID];
  __shared__ float red[256];
  const int t = threadIdx.x;

  float ss = 0.f;
  for (int i = t; i < HID; i += 256) {
    float acc = 0.f;
    for (int o = 0; o < HID; ++o) acc += W[(size_t)o * HID + i] * u[o];
    v[i] = acc;
    ss += acc * acc;
  }
  red[t] = ss; __syncthreads();
  for (int s = 128; s > 0; s >>= 1) { if (t < s) red[t] += red[t + s]; __syncthreads(); }
  const float normt = sqrtf(red[0]) + EPSV;
  __syncthreads();
  for (int i = t; i < HID; i += 256) v[i] /= normt;
  __syncthreads();

  float ssw = 0.f;
  for (int o = t; o < HID; o += 256) {
    float acc = 0.f;
    for (int i = 0; i < HID; ++i) acc += W[(size_t)o * HID + i] * v[i];
    ssw += acc * acc;
  }
  red[t] = ssw; __syncthreads();
  for (int s = 128; s > 0; s >>= 1) { if (t < s) red[t] += red[t + s]; __syncthreads(); }
  if (t == 0) {
    float nw = sqrtf(red[0]);
    float sigma = (nw * nw) / (nw + EPSV);
    inv_sigma[l] = 1.0f / sigma;
  }
}

// ---------------------------------------------------------------------------
// Conversions to bf16
// ---------------------------------------------------------------------------
__global__ __launch_bounds__(256)
void scale_w_bf16(const float* __restrict__ W, const float* __restrict__ inv_sigma,
                  __bf16* __restrict__ out, size_t per_layer, size_t n) {
  for (size_t idx = (size_t)blockIdx.x * blockDim.x + threadIdx.x; idx < n;
       idx += (size_t)gridDim.x * blockDim.x) {
    size_t layer = idx / per_layer;
    out[idx] = (__bf16)(W[idx] * inv_sigma[layer]);
  }
}

__global__ __launch_bounds__(256)
void f32_to_bf16(const float* __restrict__ in, __bf16* __restrict__ out, size_t n) {
  for (size_t idx = (size_t)blockIdx.x * blockDim.x + threadIdx.x; idx < n;
       idx += (size_t)gridDim.x * blockDim.x) {
    out[idx] = (__bf16)in[idx];
  }
}

// ---------------------------------------------------------------------------
// WMMA GEMM:  pre[M,N] = A[M,K] (bf16, row-major) * W[N,K]^T (bf16, row-major)
// Block tile 128x128, BK=32, 256 threads = 8 waves in a 2(M) x 4(N) grid;
// each wave owns a 64x32 tile = 4x2 fragments of v_wmma_f32_16x16x32_bf16.
// Tiles stream into double-buffered LDS via global_load_async_to_lds_b128,
// overlapped with the WMMA stream; one s_wait_asynccnt + barrier per K-step.
//
// MODE 0 (EMBED): outBf = bf16(acc + bias[n])
// MODE 1 (STEP) : outBf = bf16(0.5*h_old + 0.5*tanh(acc + bias[n])), per-layer
//                 pointers selected via blockIdx.z
// MODE 2 (HEAD) : outF  = acc + bias[n]
// ---------------------------------------------------------------------------
constexpr int BM = 128, BN = 128, BK = 32;
constexpr int LDS_STRIDE = 40;      // halves; padded vs 32 to dodge bank conflicts

template <int MODE>
__global__ __launch_bounds__(256)
void gemm_kernel(const __bf16* __restrict__ Abase,
                 const __bf16* __restrict__ Hsrc,    // STEP: old states, all layers
                 const __bf16* __restrict__ Wbase,
                 const float*  __restrict__ biasBase,
                 __bf16* __restrict__ outBf,
                 float*  __restrict__ outF,
                 int M, int N, int K, int lda) {
  __shared__ alignas(16) __bf16 sA[2][BM * LDS_STRIDE];
  __shared__ alignas(16) __bf16 sB[2][BN * LDS_STRIDE];

  const int t      = threadIdx.x;
  const int lane   = t & 31;
  const int waveid = t >> 5;
  const int wave_m = waveid & 1;    // 2 waves along M (64 rows each)
  const int wave_n = waveid >> 1;   // 4 waves along N (32 cols each)

  const __bf16* A    = Abase;
  const __bf16* W    = Wbase;
  const float*  bias = biasBase;
  const __bf16* hold = nullptr;
  __bf16*       obf  = outBf;

  if (MODE == 1) {
    const int layer = blockIdx.z;
    const size_t state = (size_t)M * N;               // per-layer state stride (N==K==HID)
    A    = (layer == 0) ? Abase : (Hsrc + (size_t)(layer - 1) * state);
    W    = Wbase + (size_t)layer * N * K;
    bias = biasBase + (size_t)layer * N;
    hold = Hsrc + (size_t)layer * state;
    obf  = outBf + (size_t)layer * state;
  }

  const int rowBase = blockIdx.y * BM;
  const int colBase = blockIdx.x * BN;

  // staging map: thread -> (row lr, 32B half-row lh), 4 async b128 per thread
  const int lr = t >> 1;
  const int lh = (t & 1) * 16;                        // halves
  const __bf16* gA = A + (size_t)(rowBase + lr) * lda + lh;
  const __bf16* gW = W + (size_t)(colBase + lr) * K + lh;
  const unsigned lA0 = lds_off(&sA[0][lr * LDS_STRIDE + lh]);
  const unsigned lA1 = lds_off(&sA[1][lr * LDS_STRIDE + lh]);
  const unsigned lB0 = lds_off(&sB[0][lr * LDS_STRIDE + lh]);
  const unsigned lB1 = lds_off(&sB[1][lr * LDS_STRIDE + lh]);

  v8f acc[4][2];
  #pragma unroll
  for (int f = 0; f < 4; ++f)
    #pragma unroll
    for (int nf = 0; nf < 2; ++nf)
      acc[f][nf] = (v8f)(0.0f);

  // fragment LDS offsets (contiguous k order: lane half h holds k = 16h..16h+15 for both A and B)
  const int a_klo = (lane >> 4) << 4;   // 0 or 16 (K half for A)
  const int b_klo = (lane >> 4) << 4;   // 0 or 16 (K half for B)
  const int mlane = lane & 15;

  const int nk = K / BK;

  // prologue: stream tile 0 into buffer 0
  async_copy_32B(lA0, gA);
  async_copy_32B(lB0, gW);
  wait_async0();
  __syncthreads();

  for (int kt = 0; kt < nk; ++kt) {
    const int cur = kt & 1;

    if (kt + 1 < nk) {
      // stream tile kt+1 into the other buffer while computing this one
      const int koff = (kt + 1) * BK;
      async_copy_32B(cur ? lA0 : lA1, gA + koff);
      async_copy_32B(cur ? lB0 : lB1, gW + koff);
    }

    // compute from buffer `cur`
    FragAB b[2];
    #pragma unroll
    for (int nf = 0; nf < 2; ++nf) {
      const __bf16* q = &sB[cur][(wave_n * 32 + nf * 16 + mlane) * LDS_STRIDE + b_klo];
      b[nf].h[0] = *(const v8bf*)q;
      b[nf].h[1] = *(const v8bf*)(q + 8);
    }
    #pragma unroll
    for (int f = 0; f < 4; ++f) {
      FragAB a;
      const __bf16* p = &sA[cur][(wave_m * 64 + f * 16 + mlane) * LDS_STRIDE + a_klo];
      a.h[0] = *(const v8bf*)p;
      a.h[1] = *(const v8bf*)(p + 8);
      acc[f][0] = __builtin_amdgcn_wmma_f32_16x16x32_bf16(
          false, a.v, false, b[0].v, (short)0, acc[f][0], false, false);
      acc[f][1] = __builtin_amdgcn_wmma_f32_16x16x32_bf16(
          false, a.v, false, b[1].v, (short)0, acc[f][1], false, false);
    }

    if (kt + 1 < nk) {
      wait_async0();       // tile kt+1 landed (per-wave), barrier covers all waves
      __syncthreads();
    }
  }

  // epilogue: C/D layout -> row = g + 8*(lane>=16), col = lane&15
  const int rlane = (lane >> 4) << 3;
  #pragma unroll
  for (int nf = 0; nf < 2; ++nf) {
    const int c  = colBase + wave_n * 32 + nf * 16 + mlane;
    const float bv = bias[c];
    #pragma unroll
    for (int f = 0; f < 4; ++f) {
      #pragma unroll
      for (int g = 0; g < 8; ++g) {
        const int r = rowBase + wave_m * 64 + f * 16 + rlane + g;
        const float a = acc[f][nf][g] + bv;
        if (MODE == 0) {
          obf[(size_t)r * N + c] = (__bf16)a;
        } else if (MODE == 1) {
          const float ho = (float)hold[(size_t)r * N + c];
          obf[(size_t)r * N + c] = (__bf16)((1.0f - ALPHA) * ho + ALPHA * tanhf(a));
        } else {
          outF[(size_t)r * N + c] = a;
        }
      }
    }
  }
}

// ---------------------------------------------------------------------------
// Host-side orchestration
// ---------------------------------------------------------------------------
extern "C" void kernel_launch(void* const* d_in, const int* in_sizes, int n_in,
                              void* d_out, int out_size, void* d_ws, size_t ws_size,
                              hipStream_t stream) {
  const float* x     = (const float*)d_in[0];
  const float* embW  = (const float*)d_in[1];
  const float* embB  = (const float*)d_in[2];
  const float* Ws    = (const float*)d_in[3];
  const float* bs    = (const float*)d_in[4];
  const float* us    = (const float*)d_in[5];
  const float* headW = (const float*)d_in[6];
  const float* headB = (const float*)d_in[7];
  float* out = (float*)d_out;

  char* ws = (char*)d_ws;
  size_t off = 0;
  auto alloc = [&](size_t bytes) -> void* {
    off = (off + 255) & ~(size_t)255;
    void* p = ws + off;
    off += bytes;
    return p;
  };

  float*  inv_sigma = (float*)alloc(NLAY * sizeof(float));
  __bf16* Wbf   = (__bf16*)alloc((size_t)NLAY * HID * HID * sizeof(__bf16));
  __bf16* eWbf  = (__bf16*)alloc((size_t)HID * INF * sizeof(__bf16));
  __bf16* hWbf  = (__bf16*)alloc((size_t)OUTF * HID * sizeof(__bf16));
  __bf16* xbf   = (__bf16*)alloc((size_t)BATCH * INF * sizeof(__bf16));
  __bf16* xemb  = (__bf16*)alloc((size_t)BATCH * HID * sizeof(__bf16));
  const size_t stateBytes = (size_t)NLAY * BATCH * HID * sizeof(__bf16);
  __bf16* h0 = (__bf16*)alloc(stateBytes);
  __bf16* h1 = (__bf16*)alloc(stateBytes);

  // h starts at zero (read by step 0)
  hipMemsetAsync(h0, 0, stateBytes, stream);

  // 1) spectral sigmas
  sigma_kernel<<<NLAY, 256, 0, stream>>>(Ws, us, inv_sigma);

  // 2) bf16 conversions
  scale_w_bf16<<<2048, 256, 0, stream>>>(Ws, inv_sigma, Wbf,
                                         (size_t)HID * HID,
                                         (size_t)NLAY * HID * HID);
  f32_to_bf16<<<1024, 256, 0, stream>>>(embW,  eWbf, (size_t)HID * INF);
  f32_to_bf16<<<1024, 256, 0, stream>>>(headW, hWbf, (size_t)OUTF * HID);
  f32_to_bf16<<<1024, 256, 0, stream>>>(x,     xbf,  (size_t)BATCH * INF);

  // 3) embed GEMM: x_emb = x @ embed_W^T + embed_b    [4096,2048]
  dim3 ge(HID / BN, BATCH / BM, 1);
  gemm_kernel<0><<<ge, 256, 0, stream>>>(xbf, nullptr, eWbf, embB,
                                         xemb, nullptr, BATCH, HID, INF, INF);

  // 4) 30 relaxation steps, 3 layers in parallel per step (all read old state)
  __bf16* hsrc = h0;
  __bf16* hdst = h1;
  dim3 gs(HID / BN, BATCH / BM, NLAY);
  for (int s = 0; s < STEPS; ++s) {
    gemm_kernel<1><<<gs, 256, 0, stream>>>(xemb, hsrc, Wbf, bs,
                                           hdst, nullptr, BATCH, HID, HID, HID);
    __bf16* tmp = hsrc; hsrc = hdst; hdst = tmp;
  }

  // 5) head GEMM from final state of last layer
  const __bf16* hfin = hsrc + (size_t)(NLAY - 1) * BATCH * HID;
  dim3 gh(OUTF / BN, BATCH / BM, 1);
  gemm_kernel<2><<<gh, 256, 0, stream>>>(hfin, nullptr, hWbf, headB,
                                         nullptr, out, BATCH, OUTF, HID, HID);
}
